// Decoder_41996190220903
// MI455X (gfx1250) — matrix operand order repaired and checked by reading, not compiled, not run on hardware
//
#include <hip/hip_runtime.h>
#include <hip/hip_bf16.h>

// ---------------------------------------------------------------------------
// LSTM decoder, 512 sequential steps, B=256, F=256.
// Persistent-kernel design: 16 WGs x 256 threads, weights (bf16) pinned in
// LDS, cell state in VGPRs, h ping-ponged through L2 (bf16), one grid
// barrier per step. Matrix core path: v_wmma_f32_16x16x32_bf16.
// ---------------------------------------------------------------------------

typedef __attribute__((ext_vector_type(16))) __bf16 v16bf;
typedef __attribute__((ext_vector_type(8)))  __bf16 v8bf;
typedef __attribute__((ext_vector_type(8)))  float  v8f;

#define NB 256      // batch
#define NL 128      // latent
#define NF 256      // features
#define NS 512      // seq len
#define NWG 16      // workgroups (one per 16 F-columns)
#define NTHR 256    // threads per WG (8 wave32)

union BF16x16 { v16bf v; v8bf h[2]; };

__device__ __forceinline__ float fast_sigmoid(float x) {
    return __frcp_rn(1.0f + __expf(-x));
}
__device__ __forceinline__ float fast_tanh(float x) {
    float e = __expf(2.0f * x);
    return __fdividef(e - 1.0f, e + 1.0f);
}

// Grid-wide epoch barrier (counter reset by reset_kernel each launch).
__device__ __forceinline__ void gridsync(unsigned* cnt, unsigned epoch) {
    __syncthreads();
    if (threadIdx.x == 0) {
        __threadfence();                       // release h stores to device scope
        atomicAdd(cnt, 1u);
        const unsigned target = epoch * NWG;
        while (__hip_atomic_load(cnt, __ATOMIC_RELAXED, __HIP_MEMORY_SCOPE_AGENT) < target)
            __builtin_amdgcn_s_sleep(1);
    }
    __syncthreads();
    __threadfence();                           // acquire: invalidate stale lines
}

// One GEMM accumulation pass for this wave's 2 B-tiles x 4 gates over K=256.
// A: 16x32 bf16 tiles from global (row-major NB x NF bf16 matrix `abase`).
// B: 32x16 bf16 tiles from LDS weights (64 rows x 256 K, row-major).
__device__ __forceinline__ void gemm_pass(v8f acc[2][4],
                                          const __bf16* __restrict__ abase,
                                          const __bf16* lds_w,
                                          int lane, int btile0) {
    const int nl = lane & 15;        // N column within tile / M row within tile
    const int lh = lane >> 4;        // lane half
#pragma unroll
    for (int kk = 0; kk < 8; ++kk) {
        // B fragments: lane holds N=nl, K = kk*32 + lh*16 .. +15 (contiguous)
        BF16x16 bf[4];
        const int kb = kk * 32 + lh * 16;
#pragma unroll
        for (int g = 0; g < 4; ++g) {
            const __bf16* p = lds_w + (g * 16 + nl) * NF + kb;
            bf[g].h[0] = *(const v8bf*)(p);
            bf[g].h[1] = *(const v8bf*)(p + 8);
        }
        // A fragments: lane holds M=nl, K = kk*32 + lh*16 .. +15 (contiguous)
        const int ka = kk * 32 + lh * 16;
#pragma unroll
        for (int t = 0; t < 2; ++t) {
            const __bf16* pa = abase + ((btile0 + t) * 16 + nl) * NF + ka;
            BF16x16 af;
            af.h[0] = *(const v8bf*)(pa);
            af.h[1] = *(const v8bf*)(pa + 8);
#pragma unroll
            for (int g = 0; g < 4; ++g)
                acc[t][g] = __builtin_amdgcn_wmma_f32_16x16x32_bf16(
                    false, af.v, false, bf[g].v, (short)0, acc[t][g],
                    false, false);
        }
    }
}

__global__ void reset_kernel(unsigned* cnt) { *cnt = 0u; }

__global__ void __launch_bounds__(NTHR)
lstm_decoder_kernel(const float* __restrict__ x,   const float* __restrict__ lf,
                    const float* __restrict__ Wi,  const float* __restrict__ bi,
                    const float* __restrict__ Wih, const float* __restrict__ Whh,
                    const float* __restrict__ bih, const float* __restrict__ bhh,
                    float* __restrict__ out, unsigned char* __restrict__ ws) {
    // LDS: region 0 = W_sum rows (64 x 256 bf16), region 1 = W_ih rows. 64 KB.
    __shared__ __bf16 sW[2 * 64 * NF];

    unsigned* cnt = (unsigned*)ws;
    __bf16* hbuf0 = (__bf16*)(ws + 512);
    __bf16* hbuf1 = (__bf16*)(ws + 512 + 2 * NB * NF);
    __bf16* lfm   = (__bf16*)(ws + 512 + 4 * NB * NF);      // bf16(last_feat - h0)
    float*  initf = (float*)(ws + 512 + 6 * NB * NF);       // f32 h0 (for c0)

    const int j = blockIdx.x;            // WG id -> F columns [16j, 16j+16)
    const int tid = threadIdx.x;
    const int lane = tid & 31, wave = tid >> 5;
    const int fc = j * 16;
    const int nl = lane & 15, lh = lane >> 4;
    const int btile0 = wave * 2;         // this wave's 2 B-tiles

    // ---- stage weights into LDS (once; reused for all 512 steps) ----
    for (int idx = tid; idx < 64 * NF; idx += NTHR) {
        const int rl = idx / NF, k = idx - (idx / NF) * NF;
        const int n = (rl >> 4) * NF + fc + (rl & 15);       // global gate row
        const float wih = Wih[n * NF + k];
        const float whh = Whh[n * NF + k];
        sW[rl * NF + k]           = (__bf16)(wih + whh);     // W_sum
        sW[64 * NF + rl * NF + k] = (__bf16)wih;             // W_ih
    }

    // per-lane fused bias (column fc+nl of each gate)
    float bias_g[4];
#pragma unroll
    for (int g = 0; g < 4; ++g) {
        const int n = g * NF + fc + nl;
        bias_g[g] = bih[n] + bhh[n];
    }

    // ---- init: h0 = c0 = elu(x @ Wi.T + bi) for this WG's 16 columns ----
    for (int i = tid; i < NB * 16; i += NTHR) {
        const int b = i >> 4, f = fc + (i & 15);
        float a = bi[f];
        const float* xr = x + b * NL;
        const float* wr = Wi + f * NL;
#pragma unroll 4
        for (int l = 0; l < NL; ++l) a += xr[l] * wr[l];
        const float e = (a > 0.0f) ? a : (__expf(a) - 1.0f);
        initf[b * NF + f]   = e;
        hbuf0[b * NF + f]   = (__bf16)e;
        lfm[b * NF + f]     = (__bf16)(lf[b * NF + f] - e);  // step-0 folding
    }
    __syncthreads();

    // cell state lives in VGPRs for the whole recurrence
    v8f cst[2];
#pragma unroll
    for (int t = 0; t < 2; ++t)
#pragma unroll
        for (int r = 0; r < 8; ++r) {
            const int b = (btile0 + t) * 16 + r + 8 * lh;
            cst[t][r] = initf[b * NF + fc + nl];
        }

    unsigned epoch = 1;
    gridsync(cnt, epoch);                // all WGs published h0 / lfm

    // ---- 512-step recurrence ----
    for (int s = 0; s < NS; ++s) {
        const __bf16* hread = (s & 1) ? hbuf1 : hbuf0;
        __bf16*       hwrite = (s & 1) ? hbuf0 : hbuf1;

        if (lane < 16)                   // warm our A rows (global_prefetch_b8)
            __builtin_prefetch(hread + (btile0 * 16 + lane) * NF, 0, 0);

        v8f acc[2][4];
#pragma unroll
        for (int t = 0; t < 2; ++t)
#pragma unroll
            for (int g = 0; g < 4; ++g)
#pragma unroll
                for (int r = 0; r < 8; ++r) acc[t][g][r] = bias_g[g];

        // gates = h @ W_sum.T  (+ (lf - h0) @ W_ih.T on step 0 only)
        gemm_pass(acc, hread, sW, lane, btile0);
        if (s == 0) gemm_pass(acc, lfm, sW + 64 * NF, lane, btile0);

        // LSTM cell update + stream h2 out (f32) and to next h buffer (bf16)
        float* outs = out + (size_t)s * NB * NF;
#pragma unroll
        for (int t = 0; t < 2; ++t)
#pragma unroll
            for (int r = 0; r < 8; ++r) {
                const float ig = fast_sigmoid(acc[t][0][r]);
                const float fg = fast_sigmoid(acc[t][1][r]);
                const float gg = fast_tanh(acc[t][2][r]);
                const float og = fast_sigmoid(acc[t][3][r]);
                const float c2 = fg * cst[t][r] + ig * gg;
                const float h2 = og * fast_tanh(c2);
                cst[t][r] = c2;
                const int b = (btile0 + t) * 16 + r + 8 * lh;
                const int f = fc + nl;
                outs[b * NF + f]   = h2;
                hwrite[b * NF + f] = (__bf16)h2;
            }

        gridsync(cnt, ++epoch);          // h2 visible everywhere before s+1
    }
}

extern "C" void kernel_launch(void* const* d_in, const int* in_sizes, int n_in,
                              void* d_out, int out_size, void* d_ws, size_t ws_size,
                              hipStream_t stream) {
    const float* x    = (const float*)d_in[0];
    const float* lf   = (const float*)d_in[1];
    const float* Wi   = (const float*)d_in[2];
    const float* bi   = (const float*)d_in[3];
    const float* W_ih = (const float*)d_in[4];
    const float* W_hh = (const float*)d_in[5];
    const float* b_ih = (const float*)d_in[6];
    const float* b_hh = (const float*)d_in[7];
    // d_in[8] (Wo), d_in[9] (bo): output_layer result is discarded upstream.

    reset_kernel<<<1, 1, 0, stream>>>((unsigned*)d_ws);
    lstm_decoder_kernel<<<NWG, NTHR, 0, stream>>>(
        x, lf, Wi, bi, W_ih, W_hh, b_ih, b_hh,
        (float*)d_out, (unsigned char*)d_ws);
}
